// W4A16NemotronHUniQLAttention_83416854823034
// MI455X (gfx1250) — matrix operand order repaired and checked by reading, not compiled, not run on hardware
//
#include <hip/hip_runtime.h>
#include <hip/hip_bf16.h>

// ---------------------------------------------------------------------------
// W4A16 GQA attention block for MI455X (gfx1250, wave32, WMMA 16x16x32 f16,
// async global->LDS staging)
// ---------------------------------------------------------------------------

#define S_LEN 2048
#define H_DIM 4096
#define NH    32
#define NKV   8
#define HD    128
#define GROUP 128
#define QOUT  (NH * HD)   // 4096
#define KVOUT (NKV * HD)  // 1024

typedef _Float16 half_t;
typedef half_t v4h  __attribute__((ext_vector_type(4)));
typedef half_t v8h  __attribute__((ext_vector_type(8)));
typedef half_t v16h __attribute__((ext_vector_type(16)));
typedef float  v8f  __attribute__((ext_vector_type(8)));

// Build a 16-half fragment from two 16B-aligned 8-half chunks.
__device__ __forceinline__ v16h make_frag(const half_t* p0, const half_t* p1) {
  v8h lo = *(const v8h*)p0;
  v8h hi = *(const v8h*)p1;
  v16h r;
#pragma unroll
  for (int i = 0; i < 8; ++i) { r[i] = lo[i]; r[i + 8] = hi[i]; }
  return r;
}

__device__ __forceinline__ v8f wmma_f16(v16h a, v16h b, v8f c) {
  return __builtin_amdgcn_wmma_f32_16x16x32_f16(
      /*neg_a=*/false, a, /*neg_b=*/false, b,
      /*c_mod=*/(short)0, c, /*reuse_a=*/false, /*reuse_b=*/false);
}

// Async global->LDS 16B copy (ASYNCcnt-tracked, no VGPR round trip).
// vdst VGPR holds the wave-relative LDS byte offset == low 32 bits of the
// generic pointer (aperture construction puts the LDS offset in addr[31:0]).
__device__ __forceinline__ void async_ld_b128(void* lds, const void* gaddr) {
  unsigned int loff = (unsigned int)(unsigned long long)lds;
  asm volatile("global_load_async_to_lds_b128 %0, %1, off"
               :: "v"(loff), "v"(gaddr) : "memory");
}
__device__ __forceinline__ void wait_async() {
  asm volatile("s_wait_asynccnt 0" ::: "memory");
}

// xor-shuffle within the wave via single-instruction ds_swizzle (group-of-32
// encoding: offset = xor_mask<<10 | or_mask<<5 | and_mask). Masks 1,2,4,8
// never flip lane bit 4, so reductions stay within each 16-lane half.
template <int XMASK>
__device__ __forceinline__ float swz_xor(float v) {
#if __has_builtin(__builtin_amdgcn_ds_swizzle)
  return __builtin_bit_cast(
      float, __builtin_amdgcn_ds_swizzle(__builtin_bit_cast(int, v),
                                         (XMASK << 10) | 0x1F));
#else
  return __shfl_xor(v, XMASK, 32);
#endif
}

__device__ __forceinline__ float half_red_max(float v) {
  v = fmaxf(v, swz_xor<1>(v));
  v = fmaxf(v, swz_xor<2>(v));
  v = fmaxf(v, swz_xor<4>(v));
  v = fmaxf(v, swz_xor<8>(v));
  return v;
}
__device__ __forceinline__ float half_red_sum(float v) {
  v += swz_xor<1>(v);
  v += swz_xor<2>(v);
  v += swz_xor<4>(v);
  v += swz_xor<8>(v);
  return v;
}

// ---------------------------------------------------------------------------
// fp32 -> f16 conversion of hidden states
// ---------------------------------------------------------------------------
__global__ __launch_bounds__(256) void cvt_f32_f16(const float* __restrict__ x,
                                                   half_t* __restrict__ y) {
  size_t i = ((size_t)blockIdx.x * blockDim.x + threadIdx.x) * 4;
  float4 f = *(const float4*)(x + i);
  v4h h;
  h[0] = (half_t)f.x; h[1] = (half_t)f.y; h[2] = (half_t)f.z; h[3] = (half_t)f.w;
  *(v4h*)(y + i) = h;
}

// ---------------------------------------------------------------------------
// Y[M,N] = X[M,K](f16) @ dequant(W4)[N,K]^T ; group-wise (q - zp) * sc
// WG: 256 thr = 8 waves; WG tile 64(M) x 128(N); wave tile 32x32; k-step 64.
// A tile staged via async LDS loads; B tile dequantized through VALU.
// ---------------------------------------------------------------------------
template <bool F32OUT>
__global__ __launch_bounds__(256, 1) void gemm_w4a16(
    const half_t* __restrict__ X, const int* __restrict__ qw,
    const float* __restrict__ sc, const int* __restrict__ zp,
    half_t* __restrict__ Yh, float* __restrict__ Yf, int M, int N, int K) {
  __shared__ __align__(16) half_t Asm[64 * 64];    // [m][k], stride 64
  __shared__ __align__(16) half_t Bsm[128 * 64];   // transposed: [n][k]

  const int t    = threadIdx.x;
  const int lane = t & 31;
  const int wave = t >> 5;
  const int m0   = blockIdx.x * 64;
  const int n0   = blockIdx.y * 128;
  const int mloc = (wave >> 2) * 32;   // 0 or 32
  const int nw   = (wave & 3) * 32;    // 0,32,64,96
  const int l15  = lane & 15;
  const int abase = (lane < 16) ? 0 : 16;   // 16-bit A fragment K base
  const int bbase = (lane < 16) ? 0 : 16;   // 16-bit B fragment K base
  const int ngroups = K / GROUP;

  v8f acc[2][2] = {};

  for (int k0 = 0; k0 < K; k0 += 64) {
    // ---- stage A tile 64x64: async global->LDS, 2 x b128 per thread ----
#pragma unroll
    for (int i = 0; i < 2; ++i) {
      const int idx  = t + i * 256;
      const int m    = idx >> 3;
      const int koff = (idx & 7) * 8;
      async_ld_b128(&Asm[m * 64 + koff],
                    X + (size_t)(m0 + m) * K + k0 + koff);
    }
    // ---- stage B tile, dequant int4(int32) -> f16, k-major ----
#pragma unroll
    for (int i = 0; i < 2; ++i) {
      const int idx  = t + i * 256;
      const int n    = idx >> 2;
      const int koff = (idx & 3) * 16;
      const int o    = n0 + n;
      const int g    = (k0 + koff) / GROUP;   // 16-chunk never crosses a group
      const float s  = sc[(size_t)o * ngroups + g];
      const float z  = (float)zp[(size_t)o * ngroups + g];
      const int* q   = qw + (size_t)o * K + k0 + koff;
      __builtin_prefetch(q + 64, 0, 1);       // global_prefetch_b8 next k-step
      v8h d0, d1;
#pragma unroll
      for (int j = 0; j < 8; ++j) {
        d0[j] = (half_t)(((float)q[j] - z) * s);
        d1[j] = (half_t)(((float)q[j + 8] - z) * s);
      }
      half_t* dst = &Bsm[n * 64 + koff];
      *(v8h*)dst       = d0;
      *(v8h*)(dst + 8) = d1;
    }
    wait_async();
    __syncthreads();

#pragma unroll
    for (int kk = 0; kk < 64; kk += 32) {
      const half_t* ar0 = &Asm[(mloc + l15) * 64 + kk + abase];
      const half_t* ar1 = &Asm[(mloc + 16 + l15) * 64 + kk + abase];
      v16h a0 = make_frag(ar0, ar0 + 8);
      v16h a1 = make_frag(ar1, ar1 + 8);
      const half_t* br0 = &Bsm[(nw + l15) * 64 + kk + bbase];
      const half_t* br1 = &Bsm[(nw + 16 + l15) * 64 + kk + bbase];
      v16h b0 = make_frag(br0, br0 + 8);
      v16h b1 = make_frag(br1, br1 + 8);

      acc[0][0] = wmma_f16(a0, b0, acc[0][0]);
      acc[0][1] = wmma_f16(a0, b1, acc[0][1]);
      acc[1][0] = wmma_f16(a1, b0, acc[1][0]);
      acc[1][1] = wmma_f16(a1, b1, acc[1][1]);
    }
    __syncthreads();
  }

  // ---- epilogue: C layout VGPR r <-> row r (lanes 0-15) / 8+r (lanes 16-31)
  const int mhalf = (lane < 16) ? 0 : 8;
#pragma unroll
  for (int i = 0; i < 2; ++i)
#pragma unroll
    for (int j = 0; j < 2; ++j)
#pragma unroll
      for (int r = 0; r < 8; ++r) {
        const int row = m0 + mloc + i * 16 + r + mhalf;
        const int col = n0 + nw + j * 16 + l15;
        if (F32OUT) Yf[(size_t)row * N + col] = acc[i][j][r];
        else        Yh[(size_t)row * N + col] = (half_t)acc[i][j][r];
      }
}

// ---------------------------------------------------------------------------
// Causal GQA flash attention.  grid = (S/128, NH), block = 256 (8 waves).
// Wave w owns 16 query rows; WG covers 128 rows of one head.
// Q/K tiles staged by async LDS loads; V transposed through VALU.
// ---------------------------------------------------------------------------
__global__ __launch_bounds__(256, 1) void flash_attn(
    const half_t* __restrict__ Q, const half_t* __restrict__ Kv,
    const half_t* __restrict__ Vv, half_t* __restrict__ Oout) {
  __shared__ __align__(16) half_t Qsm[128 * HD];      // 32 KB [qrow][hd]
  __shared__ __align__(16) half_t Ksm[32 * HD];       //  8 KB [key][hd]
  __shared__ __align__(16) half_t Vsm[HD * 32];       //  8 KB transposed
  __shared__ __align__(16) half_t Psm[8][16 * 32];    //  8 KB per-wave P

  const int t    = threadIdx.x;
  const int lane = t & 31;
  const int wave = t >> 5;
  const int head = blockIdx.y;
  const int kvh  = head >> 2;          // GQA: 4 q-heads per kv-head
  const int q0   = blockIdx.x * 128;
  const int qw0  = q0 + wave * 16;     // this wave's first query row
  const int l15  = lane & 15;
  const int abase = (lane < 16) ? 0 : 16;
  const int bbase = (lane < 16) ? 0 : 16;
  const int mhalf = (lane < 16) ? 0 : 8;
  const float scale = 0.08838834764831845f;  // 1/sqrt(128)

  // ---- stage WG's Q tile 128x128 once (async) ----
#pragma unroll
  for (int i = 0; i < 8; ++i) {
    const int idx  = t + i * 256;
    const int row  = idx >> 4;
    const int koff = (idx & 15) * 8;
    async_ld_b128(&Qsm[row * HD + koff],
                  Q + (size_t)(q0 + row) * QOUT + head * HD + koff);
  }

  v8f   oacc[8] = {};
  float m_run[8], l_run[8];
#pragma unroll
  for (int r = 0; r < 8; ++r) { m_run[r] = -3.0e38f; l_run[r] = 0.0f; }
  wait_async();
  __syncthreads();

  const int jmax = (q0 + 127) / 32;
  for (int j = 0; j <= jmax; ++j) {
    const int kb = j * 32;

    // ---- stage K tile [32 keys][128] (async) + V^T tile [128][32] ----
    {
      const int key  = t >> 3;
      const int hoff = (t & 7) * 16;
      const half_t* ksrc = Kv + (size_t)(kb + key) * KVOUT + kvh * HD + hoff;
      async_ld_b128(&Ksm[key * HD + hoff],     ksrc);
      async_ld_b128(&Ksm[key * HD + hoff + 8], ksrc + 8);
      const half_t* vsrc = Vv + (size_t)(kb + key) * KVOUT + kvh * HD + hoff;
#pragma unroll
      for (int i = 0; i < 16; ++i)
        Vsm[(hoff + i) * 32 + key] = vsrc[i];
    }
    wait_async();
    __syncthreads();

    if (kb <= qw0 + 15) {  // wave has at least one unmasked key in tile
      // ---- scores = Q @ K^T : 16 x 32 ----
      v8f s0 = {}, s1 = {};
#pragma unroll
      for (int c = 0; c < 4; ++c) {
        const half_t* qp = &Qsm[(wave * 16 + l15) * HD + c * 32 + abase];
        v16h aq = make_frag(qp, qp + 8);
        const half_t* k0p = &Ksm[l15 * HD + c * 32 + bbase];
        const half_t* k1p = &Ksm[(16 + l15) * HD + c * 32 + bbase];
        s0 = wmma_f16(aq, make_frag(k0p, k0p + 8), s0);
        s1 = wmma_f16(aq, make_frag(k1p, k1p + 8), s1);
      }

      // ---- online softmax (row reductions across 16-lane halves) ----
#pragma unroll
      for (int r = 0; r < 8; ++r) {
        const int qg = qw0 + r + mhalf;          // global query row
        float v0 = s0[r] * scale;
        float v1 = s1[r] * scale;
        if (kb + l15 > qg)      v0 = -1.0e9f;
        if (kb + 16 + l15 > qg) v1 = -1.0e9f;

        const float rm   = half_red_max(fmaxf(v0, v1));
        const float mnew = fmaxf(m_run[r], rm);
        const float corr = __expf(m_run[r] - mnew);
        const float p0 = __expf(v0 - mnew);
        const float p1 = __expf(v1 - mnew);
        const float ls = half_red_sum(p0 + p1);
        l_run[r] = l_run[r] * corr + ls;
        m_run[r] = mnew;
#pragma unroll
        for (int c = 0; c < 8; ++c) oacc[c][r] *= corr;

        const int m = r + mhalf;
        Psm[wave][m * 32 + l15]      = (half_t)p0;
        Psm[wave][m * 32 + 16 + l15] = (half_t)p1;
      }

      // ---- O += P @ V : reshape P through LDS into an A-fragment ----
      const half_t* pr = &Psm[wave][l15 * 32 + abase];
      v16h ap = make_frag(pr, pr + 8);
#pragma unroll
      for (int c = 0; c < 8; ++c) {
        const half_t* vp = &Vsm[(c * 16 + l15) * 32 + bbase];
        oacc[c] = wmma_f16(ap, make_frag(vp, vp + 8), oacc[c]);
      }
    }
    __syncthreads();
  }

  // ---- normalize and write out [S, NH*HD] f16 ----
  float inv[8];
#pragma unroll
  for (int r = 0; r < 8; ++r) inv[r] = 1.0f / l_run[r];
#pragma unroll
  for (int c = 0; c < 8; ++c)
#pragma unroll
    for (int r = 0; r < 8; ++r) {
      const int row = qw0 + r + mhalf;
      const int col = head * HD + c * 16 + l15;
      Oout[(size_t)row * QOUT + col] = (half_t)(oacc[c][r] * inv[r]);
    }
}

// ---------------------------------------------------------------------------
extern "C" void kernel_launch(void* const* d_in, const int* in_sizes, int n_in,
                              void* d_out, int out_size, void* d_ws, size_t ws_size,
                              hipStream_t stream) {
  (void)in_sizes; (void)n_in; (void)out_size; (void)ws_size;
  const float* x    = (const float*)d_in[0];
  const int*   q_qw = (const int*)d_in[1];
  const float* q_sc = (const float*)d_in[2];
  const int*   q_zp = (const int*)d_in[3];
  const int*   k_qw = (const int*)d_in[4];
  const float* k_sc = (const float*)d_in[5];
  const int*   k_zp = (const int*)d_in[6];
  const int*   v_qw = (const int*)d_in[7];
  const float* v_sc = (const float*)d_in[8];
  const int*   v_zp = (const int*)d_in[9];
  const int*   o_qw = (const int*)d_in[10];
  const float* o_sc = (const float*)d_in[11];
  const int*   o_zp = (const int*)d_in[12];

  char* ws = (char*)d_ws;
  half_t* xh = (half_t*)(ws);                          // 16 MB
  half_t* qh = (half_t*)(ws + ((size_t)16 << 20));     // 16 MB
  half_t* kh = (half_t*)(ws + ((size_t)32 << 20));     //  4 MB
  half_t* vh = (half_t*)(ws + ((size_t)36 << 20));     //  4 MB
  half_t* ah = (half_t*)(ws + ((size_t)40 << 20));     // 16 MB

  // 1) hidden f32 -> f16
  cvt_f32_f16<<<(S_LEN * H_DIM) / (256 * 4), 256, 0, stream>>>(x, xh);

  // 2) Q/K/V projections (W4 dequant fused into WMMA GEMM)
  gemm_w4a16<false><<<dim3(S_LEN / 64, QOUT / 128), 256, 0, stream>>>(
      xh, q_qw, q_sc, q_zp, qh, nullptr, S_LEN, QOUT, H_DIM);
  gemm_w4a16<false><<<dim3(S_LEN / 64, KVOUT / 128), 256, 0, stream>>>(
      xh, k_qw, k_sc, k_zp, kh, nullptr, S_LEN, KVOUT, H_DIM);
  gemm_w4a16<false><<<dim3(S_LEN / 64, KVOUT / 128), 256, 0, stream>>>(
      xh, v_qw, v_sc, v_zp, vh, nullptr, S_LEN, KVOUT, H_DIM);

  // 3) causal GQA flash attention
  flash_attn<<<dim3(S_LEN / 128, NH), 256, 0, stream>>>(qh, kh, vh, ah);

  // 4) O projection, f32 epilogue straight into d_out
  gemm_w4a16<true><<<dim3(S_LEN / 64, H_DIM / 128), 256, 0, stream>>>(
      ah, o_qw, o_sc, o_zp, nullptr, (float*)d_out, S_LEN, H_DIM, H_DIM);
}
